// Attention_7430293422579
// MI455X (gfx1250) — compile-verified, matrix operand order checked by reading, not run on hardware
//
#include <hip/hip_runtime.h>
#include <hip/hip_bf16.h>
#include <stdint.h>

// ---------------------------------------------------------------------------
// Problem constants (fixed by the reference): B=2, T=2048, D_MODEL=2048,
// H=16, D=128.  All GEMMs are 4096 x 2048 x 2048.
// ---------------------------------------------------------------------------
#define BB 2
#define TT 2048
#define DM 2048
#define HH 16
#define DH 128
#define MROWS (BB * TT)   // 4096

typedef __attribute__((ext_vector_type(16))) _Float16 v16h;
typedef __attribute__((ext_vector_type(8)))  float    v8f;
typedef int v4i_blk __attribute__((vector_size(16)));   // b128 payload type

// ---- gfx1250 async global->LDS copy (guarded; falls back to sync copy) ----
#if defined(__has_builtin)
#if __has_builtin(__builtin_amdgcn_global_load_async_to_lds_b128)
#define HAVE_ASYNC_LDS 1
#endif
#endif

__device__ __forceinline__ void copy16B_to_lds(const _Float16* g, _Float16* l) {
#ifdef HAVE_ASYNC_LDS
  __builtin_amdgcn_global_load_async_to_lds_b128(
      (__attribute__((address_space(1))) v4i_blk*)(g),
      (__attribute__((address_space(3))) v4i_blk*)(l), 0, 0);
#else
  *(uint4*)l = *(const uint4*)g;
#endif
}

__device__ __forceinline__ void wait_async_copies() {
#ifdef HAVE_ASYNC_LDS
#if __has_builtin(__builtin_amdgcn_s_wait_asynccnt)
  __builtin_amdgcn_s_wait_asynccnt(0);
#else
  asm volatile("s_wait_asynccnt 0" ::: "memory");
#endif
#endif
}

__device__ __forceinline__ v8f wmma_f16(v16h a, v16h b, v8f c) {
  // D = A(16x32 f16) * B(32x16 f16) + C(16x16 f32)
  return __builtin_amdgcn_wmma_f32_16x16x32_f16(
      /*neg_a=*/false, a, /*neg_b=*/false, b,
      /*c_mod=*/(short)0, c, /*reuse_a=*/false, /*reuse_b=*/false);
}

// Load a 16x32 f16 fragment from an LDS image laid out [row][k] with `stride`
// halves per row.  Works for A fragments (row = M) and for B fragments when
// the B tile is stored k-contiguous per output column (row = N).
// CDNA5 16-bit A layout: lane = (m | h<<4), VGPR v holds
// K = 2*(v&3) + 16*(v>>2) + 8*h  and  K+1.   -> two ds_load_b128 per lane.
__device__ __forceinline__ v16h load_frag16x32(const _Float16* p, int stride) {
  const int lane = threadIdx.x & 31;
  const int m = lane & 15;
  const int h = lane >> 4;
  v16h f;
#pragma unroll
  for (int v = 0; v < 8; ++v) {
    const int k0 = ((v & 3) * 2) + ((v >> 2) * 16) + h * 8;
    const _Float16* q = p + m * stride + k0;
    f[2 * v]     = q[0];
    f[2 * v + 1] = q[1];
  }
  return f;
}

// ---------------------------------------------------------------------------
// fp32 -> f16 elementwise conversion (x activations).
// ---------------------------------------------------------------------------
__global__ __launch_bounds__(256) void convert_f16(const float* __restrict__ src,
                                                   _Float16* __restrict__ dst) {
  const size_t i = ((size_t)blockIdx.x * 256 + threadIdx.x) * 4;
  const float4 v = *(const float4*)(src + i);
  dst[i + 0] = (_Float16)v.x;
  dst[i + 1] = (_Float16)v.y;
  dst[i + 2] = (_Float16)v.z;
  dst[i + 3] = (_Float16)v.w;
}

// ---------------------------------------------------------------------------
// fp32 (K x N) -> f16 transposed (N x K) via LDS 32x32 tiles.
// grid = (N/32, K/32), 256 threads.
// ---------------------------------------------------------------------------
__global__ __launch_bounds__(256) void convert_transpose(const float* __restrict__ W,
                                                         _Float16* __restrict__ Wt) {
  __shared__ float tile[32][33];
  const int c  = threadIdx.x & 31;
  const int r0 = threadIdx.x >> 5;  // 0..7
  const int n0 = blockIdx.x * 32;
  const int k0 = blockIdx.y * 32;
#pragma unroll
  for (int rr = 0; rr < 32; rr += 8)
    tile[r0 + rr][c] = W[(size_t)(k0 + r0 + rr) * DM + n0 + c];
  __syncthreads();
#pragma unroll
  for (int rr = 0; rr < 32; rr += 8)
    Wt[(size_t)(n0 + r0 + rr) * DM + k0 + c] = (_Float16)tile[c][r0 + rr];
}

// ---------------------------------------------------------------------------
// WMMA GEMM:  C(4096x2048) = A(4096x2048 f16) * Bt^T  (Bt is N x K f16)
//   EPI 0 -> store f16 into [b][h][t][d] tensor (QKV projections)
//   EPI 1 -> store f32 row-major (final output projection)
// Block tile 128x128, BK=32, 256 threads = 8 waves, wave tile 32x64 (2x4).
// ---------------------------------------------------------------------------
template <int EPI>
__global__ __launch_bounds__(256) void gemm_wmma(const _Float16* __restrict__ A,
                                                 const _Float16* __restrict__ Bt,
                                                 void* __restrict__ dst) {
  __shared__ _Float16 sA[128 * 40];  // [m][k], padded stride 40 (80B = 5*16B)
  __shared__ _Float16 sB[128 * 40];  // [n][k], padded stride 40

  const int tid  = threadIdx.x;
  const int wave = tid >> 5;
  const int lane = tid & 31;
  const int wRow = wave & 3;   // 4 * 32 = 128 rows
  const int wCol = wave >> 2;  // 2 * 64 = 128 cols
  const int m0 = blockIdx.y * 128;
  const int n0 = blockIdx.x * 128;

  v8f zero = {};
  v8f acc[2][4];
#pragma unroll
  for (int i = 0; i < 2; ++i)
#pragma unroll
    for (int j = 0; j < 4; ++j) acc[i][j] = zero;

  for (int kb = 0; kb < DM; kb += 32) {
    __syncthreads();
    // Stage A and B tiles: 128 rows x 32 halves = 64B/row, 16B per transfer.
    // 512 transfers per tile, 256 threads -> 2 each per tile.
#pragma unroll
    for (int i = tid; i < 512; i += 256) {
      const int r = i >> 2, seg = (i & 3) * 8;
      copy16B_to_lds(A  + (size_t)(m0 + r) * DM + kb + seg, sA + r * 40 + seg);
      copy16B_to_lds(Bt + (size_t)(n0 + r) * DM + kb + seg, sB + r * 40 + seg);
    }
    wait_async_copies();
    __syncthreads();

    v16h af[2], bf[4];
#pragma unroll
    for (int fm = 0; fm < 2; ++fm)
      af[fm] = load_frag16x32(sA + (wRow * 32 + fm * 16) * 40, 40);
#pragma unroll
    for (int fn = 0; fn < 4; ++fn)
      bf[fn] = load_frag16x32(sB + (wCol * 64 + fn * 16) * 40, 40);
#pragma unroll
    for (int fm = 0; fm < 2; ++fm)
#pragma unroll
      for (int fn = 0; fn < 4; ++fn)
        acc[fm][fn] = wmma_f16(af[fm], bf[fn], acc[fm][fn]);
  }

  // Epilogue.  C layout: lane (n = lane&15, h = lane>>4), VGPR r -> row r+8h.
  const int n = lane & 15;
  const int h = lane >> 4;
#pragma unroll
  for (int fm = 0; fm < 2; ++fm)
#pragma unroll
    for (int fn = 0; fn < 4; ++fn)
#pragma unroll
      for (int r = 0; r < 8; ++r) {
        const int gr = m0 + wRow * 32 + fm * 16 + r + 8 * h;
        const int gc = n0 + wCol * 64 + fn * 16 + n;
        const float val = acc[fm][fn][r];
        if (EPI == 0) {
          const int b = gr >> 11, t = gr & (TT - 1);
          const int hh = gc >> 7, d = gc & (DH - 1);
          ((_Float16*)dst)[(((size_t)b * HH + hh) * TT + t) * DH + d] =
              (_Float16)val;
        } else {
          ((float*)dst)[(size_t)gr * DM + gc] = val;
        }
      }
}

// ---------------------------------------------------------------------------
// RoPE on Q and K, layout [b*h][t][d], f16 storage, fp32 math.
// ---------------------------------------------------------------------------
__global__ __launch_bounds__(256) void rope_kernel(_Float16* __restrict__ q,
                                                   _Float16* __restrict__ k,
                                                   const float* __restrict__ cosT,
                                                   const float* __restrict__ sinT) {
  const int idx = blockIdx.x * 256 + threadIdx.x;  // B*H*T*64 total
  const int d  = idx & 63;
  const int t  = (idx >> 6) & (TT - 1);
  const int bh = idx >> 17;
  const size_t base = ((size_t)bh * TT + t) * DH;
  const float c = cosT[t * 64 + d];
  const float s = sinT[t * 64 + d];
  float q1 = (float)q[base + d], q2 = (float)q[base + d + 64];
  q[base + d]      = (_Float16)(q1 * c - q2 * s);
  q[base + d + 64] = (_Float16)(q1 * s + q2 * c);
  float k1 = (float)k[base + d], k2 = (float)k[base + d + 64];
  k[base + d]      = (_Float16)(k1 * c - k2 * s);
  k[base + d + 64] = (_Float16)(k1 * s + k2 * c);
}

// ---------------------------------------------------------------------------
// Flash attention (causal).  grid = (T/64, B*H), 128 threads = 4 waves.
// Wave w owns 16 query rows.  K tiles of 32 keys streamed through LDS.
// ---------------------------------------------------------------------------
__global__ __launch_bounds__(128) void attn_kernel(const _Float16* __restrict__ qp,
                                                   const _Float16* __restrict__ kp,
                                                   const _Float16* __restrict__ vp,
                                                   _Float16* __restrict__ outp) {
  __shared__ _Float16 sQ[64 * 136];   // [qrow][d], 272B row (17*16B)
  __shared__ _Float16 sK[32 * 136];   // [key][d]
  __shared__ _Float16 sVT[128 * 40];  // [d][key]   (V transposed)
  __shared__ _Float16 sP[4 * 16 * 40];// per-wave P tile [m][key]

  const int tid  = threadIdx.x;
  const int wave = tid >> 5;
  const int lane = tid & 31;
  const int n = lane & 15;
  const int h = lane >> 4;
  const int qTile = blockIdx.x;
  const int bh    = blockIdx.y;
  const int qBase = qTile * 64;

  // Load Q tile (64 x 128 halves): 16 x 16B per row, 1024 transfers / 128 thr.
  {
    const _Float16* qg = qp + ((size_t)bh * TT + qBase) * DH;
#pragma unroll
    for (int i = tid; i < 1024; i += 128) {
      const int r = i >> 4, seg = (i & 15) * 8;
      copy16B_to_lds(qg + (size_t)r * DH + seg, sQ + r * 136 + seg);
    }
  }

  v8f zero = {};
  float mrow[8], lrow[8];
  v8f o[8];
#pragma unroll
  for (int r = 0; r < 8; ++r) { mrow[r] = -1e30f; lrow[r] = 0.f; }
#pragma unroll
  for (int df = 0; df < 8; ++df) o[df] = zero;

  const float scale = 0.08838834764831845f;  // 1/sqrt(128)
  const int nTiles = qTile * 2 + 2;          // causal: keys < qBase+64

  for (int kt = 0; kt < nTiles; ++kt) {
    const int kBase = kt * 32;
    __syncthreads();
    // Stage K tile 32 x 128 halves (async b128).
    {
      const _Float16* kg = kp + ((size_t)bh * TT + kBase) * DH;
#pragma unroll
      for (int i = tid; i < 512; i += 128) {
        const int r = i >> 4, seg = (i & 15) * 8;
        copy16B_to_lds(kg + (size_t)r * DH + seg, sK + r * 136 + seg);
      }
    }
    // Stage V transposed: sVT[d][key]  (element transpose, plain DS stores).
    {
      const _Float16* vg = vp + ((size_t)bh * TT + kBase) * DH;
#pragma unroll
      for (int i = tid; i < 32 * 128; i += 128) {
        const int d = i & 127, kk = i >> 7;
        sVT[d * 40 + kk] = vg[(size_t)kk * DH + d];
      }
    }
    wait_async_copies();  // covers Q (first iteration) and K
    __syncthreads();

    // S(16x32) = Q_w (16x128) * K^T : 2 key-frags x 4 d-chunks
    v8f s[2] = {zero, zero};
#pragma unroll
    for (int kc = 0; kc < 4; ++kc) {
      v16h a = load_frag16x32(sQ + (wave * 16) * 136 + kc * 32, 136);
#pragma unroll
      for (int fn = 0; fn < 2; ++fn) {
        v16h b = load_frag16x32(sK + (fn * 16) * 136 + kc * 32, 136);
        s[fn] = wmma_f16(a, b, s[fn]);
      }
    }

    // Scale + causal mask.
#pragma unroll
    for (int fn = 0; fn < 2; ++fn)
#pragma unroll
      for (int r = 0; r < 8; ++r) {
        const int row = qBase + wave * 16 + r + 8 * h;
        const int key = kBase + fn * 16 + n;
        const float v = s[fn][r] * scale;
        s[fn][r] = (key <= row) ? v : -1e30f;
      }

    // Online softmax: per-lane rows r+8h, reduce across the 16-lane half
    // (xor 1,2,4,8 never crosses bit 4, so halves stay separate).
    float mnew[8], alpha[8];
#pragma unroll
    for (int r = 0; r < 8; ++r) {
      float mx = fmaxf(s[0][r], s[1][r]);
#pragma unroll
      for (int off = 8; off >= 1; off >>= 1)
        mx = fmaxf(mx, __shfl_xor(mx, off, 32));
      mnew[r]  = fmaxf(mrow[r], mx);
      alpha[r] = __expf(mrow[r] - mnew[r]);
      mrow[r]  = mnew[r];
    }
#pragma unroll
    for (int r = 0; r < 8; ++r) {
      const float p0 = __expf(s[0][r] - mnew[r]);
      const float p1 = __expf(s[1][r] - mnew[r]);
      s[0][r] = p0; s[1][r] = p1;
      float sum = p0 + p1;
#pragma unroll
      for (int off = 8; off >= 1; off >>= 1) sum += __shfl_xor(sum, off, 32);
      lrow[r] = lrow[r] * alpha[r] + sum;
    }

    // Re-layout P: C-frag -> LDS [m][key] so it can be reloaded as an A-frag.
    _Float16* pw = sP + wave * 16 * 40;
#pragma unroll
    for (int fn = 0; fn < 2; ++fn)
#pragma unroll
      for (int r = 0; r < 8; ++r)
        pw[(r + 8 * h) * 40 + fn * 16 + n] = (_Float16)s[fn][r];

    // Rescale O accumulators.
#pragma unroll
    for (int df = 0; df < 8; ++df)
#pragma unroll
      for (int r = 0; r < 8; ++r) o[df][r] *= alpha[r];

    // O(16x128) += P(16x32) * V(32x128): one A-frag, 8 B-frags (d columns).
    v16h ap = load_frag16x32(pw, 40);
#pragma unroll
    for (int df = 0; df < 8; ++df) {
      v16h bv = load_frag16x32(sVT + (df * 16) * 40, 40);
      o[df] = wmma_f16(ap, bv, o[df]);
    }
  }

  // Normalize and store to [b][t][h*128+d] (f16), feeding the out-projection.
  const int b    = bh >> 4;
  const int head = bh & 15;
#pragma unroll
  for (int df = 0; df < 8; ++df)
#pragma unroll
    for (int r = 0; r < 8; ++r) {
      const int row = qBase + wave * 16 + r + 8 * h;
      const int col = head * DH + df * 16 + n;
      const float val = o[df][r] / lrow[r];
      outp[((size_t)b * TT + row) * DM + col] = (_Float16)val;
    }
}

// ---------------------------------------------------------------------------
// Launch.  Inputs: x, mask(ignored; causal hardcoded), cos, sin, Wq,Wk,Wv,Wo.
// Output: (B,T,D_MODEL) fp32.
// Workspace (f16): xh 16MiB | wt 8MiB (reused per weight, stream-serial) |
//                  q,k,v,attn 4 x 16MiB   -> ~88 MiB total.
// ---------------------------------------------------------------------------
extern "C" void kernel_launch(void* const* d_in, const int* in_sizes, int n_in,
                              void* d_out, int out_size, void* d_ws, size_t ws_size,
                              hipStream_t stream) {
  const float* x    = (const float*)d_in[0];
  const float* cosT = (const float*)d_in[2];
  const float* sinT = (const float*)d_in[3];
  const float* Wq   = (const float*)d_in[4];
  const float* Wk   = (const float*)d_in[5];
  const float* Wv   = (const float*)d_in[6];
  const float* Wo   = (const float*)d_in[7];

  const size_t TEN  = (size_t)BB * HH * TT * DH;  // 8,388,608
  const size_t WTEN = (size_t)DM * DM;            // 4,194,304
  _Float16* xh  = (_Float16*)d_ws;
  _Float16* wt  = xh + TEN;
  _Float16* qws = wt + WTEN;
  _Float16* kws = qws + TEN;
  _Float16* vws = kws + TEN;
  _Float16* aws = vws + TEN;

  const dim3 gT(DM / 32, DM / 32);      // transpose grid (64, 64)
  const dim3 gG(DM / 128, MROWS / 128); // GEMM grid (16, 32)

  convert_f16<<<TEN / (256 * 4), 256, 0, stream>>>(x, xh);

  convert_transpose<<<gT, 256, 0, stream>>>(Wq, wt);
  gemm_wmma<0><<<gG, 256, 0, stream>>>(xh, wt, qws);
  convert_transpose<<<gT, 256, 0, stream>>>(Wk, wt);
  gemm_wmma<0><<<gG, 256, 0, stream>>>(xh, wt, kws);
  convert_transpose<<<gT, 256, 0, stream>>>(Wv, wt);
  gemm_wmma<0><<<gG, 256, 0, stream>>>(xh, wt, vws);

  rope_kernel<<<(BB * HH * TT * 64) / 256, 256, 0, stream>>>(qws, kws, cosT, sinT);

  const dim3 ga(TT / 64, BB * HH);  // (32, 32)
  attn_kernel<<<ga, 128, 0, stream>>>(qws, kws, vws, aws);

  convert_transpose<<<gT, 256, 0, stream>>>(Wo, wt);
  gemm_wmma<1><<<gG, 256, 0, stream>>>(aws, wt, d_out);
}
